// DynamicsRotamer_71640054497689
// MI455X (gfx1250) — matrix operand order repaired and checked by reading, not compiled, not run on hardware
//
#include <hip/hip_runtime.h>
#include <hip/hip_bf16.h>
#include <math.h>

// ---------------------------------------------------------------------------
// EGNN forward on MI455X (gfx1250), wave32 + v_wmma_f32_16x16x32_f16.
// Edge-MLP GEMM1 is decomposed into node-level projections; the only large
// GEMM (silu(pre1) @ We2, 204800x128x128 per layer) runs fused in k_edge with
// the A operand synthesized on the fly in WMMA register layout.
// Edge grid is split (bs, i-tile, j-half) -> 160 blocks x 5 waves to fill the
// WGPs; cross-block agg/xout partials combined with global_atomic_add_f32.
// ---------------------------------------------------------------------------

typedef __attribute__((ext_vector_type(16))) _Float16 v16h;
typedef __attribute__((ext_vector_type(8)))  float    v8f;

#define WMMA16(a,b,c) __builtin_amdgcn_wmma_f32_16x16x32_f16( \
    false,(a),false,(b),(short)0,(c),false,false)

static constexpr int Bb  = 2, Ss = 4, Nn = 160, HID = 128;
static constexpr int BS  = Bb * Ss;     // 8 graphs
static constexpr int M   = BS * Nn;     // 1280 node rows
static constexpr int MT  = M / 16;      // 80 M-tiles
static constexpr int NAT = 43, NAA = 20, NRES = 15;
static constexpr int EDGE_THREADS = 160;  // 5 waves, one 16-wide j-tile each

__device__ __forceinline__ float siluf(float x){ return x / (1.f + __expf(-x)); }

union Frag16 { _Float16 h[16]; uint4 q[2]; v16h v; };

// ---------------------------------------------------------------------------
// Embedding: one-hot(atom)|one-hot(aa)|one-hot(res), zero-padded 78 -> 128.
__global__ void k_init(const int* __restrict__ atoms, const int* __restrict__ frag,
                       const int* __restrict__ apos, float* __restrict__ h){
  int idx = blockIdx.x * blockDim.x + threadIdx.x;
  if (idx >= Bb * Nn) return;
  int b = idx / Nn, n = idx % Nn;
  int at  = atoms[b * Nn + n];
  int pos = apos [b * Nn + n] - 1;
  int aa  = frag [b * NRES + pos];
  for (int s = 0; s < Ss; ++s){
    float* hr = h + ((size_t)((b * Ss + s) * Nn) + n) * HID;
    for (int c = 0; c < HID; ++c) hr[c] = 0.f;
    hr[at] = 1.f; hr[NAT + aa] = 1.f; hr[NAT + NAA + pos] = 1.f;
  }
}

__global__ void k_copyx(const float* __restrict__ x, float* __restrict__ xc){
  int i = blockIdx.x * blockDim.x + threadIdx.x;
  if (i < M * 3) xc[i] = x[i];
}

__global__ void k_zero(float* __restrict__ p, int n){
  int i = blockIdx.x * blockDim.x + threadIdx.x;
  if (i < n) p[i] = 0.f;
}

// ---------------------------------------------------------------------------
// Zero-pad weight rows into a [Kpad][128] f32 buffer (two source segments).
__global__ void k_pad2(const float* __restrict__ s1, int r1, int d1,
                       const float* __restrict__ s2, int r2, int d2,
                       float* __restrict__ dst, int Kpad){
  int idx = blockIdx.x * blockDim.x + threadIdx.x;
  if (idx >= Kpad * 128) return;
  int row = idx >> 7, col = idx & 127;
  float v = 0.f;
  if (row >= d1 && row < d1 + r1) v = s1[(size_t)(row - d1) * 128 + col];
  if (s2 && row >= d2 && row < d2 + r2) v = s2[(size_t)(row - d2) * 128 + col];
  dst[idx] = v;
}

// Pack row-major f32 [KC*32][128] into WMMA B-operand f16 layout:
// lane L holds col = L%16, K = kc*32 + (L<16?0:8) + (p<8 ? p : p+8).
__global__ void k_pack_b(const float* __restrict__ src, int KC, _Float16* __restrict__ dst){
  int idx = blockIdx.x * blockDim.x + threadIdx.x;
  if (idx >= KC * 4096) return;
  int p = idx & 15, lane = (idx >> 4) & 31, nt = (idx >> 9) & 7, kc = idx >> 12;
  int K = kc * 32 + ((lane & 16) ? 8 : 0) + (p < 8 ? p : p + 8);
  dst[idx] = (_Float16)src[(size_t)K * 128 + nt * 16 + (lane & 15)];
}

// Pack row-major f32 [mtiles*16][128] into WMMA A-operand f16 layout:
// lane L holds row = L%16, K = kc*32 + base + (p<8 ? p : p+8), base = (L<16?0:8).
__global__ void k_pack_a(const float* __restrict__ src, int mtiles,
                         _Float16* __restrict__ dst, int kcSlots, int kcOff){
  int idx = blockIdx.x * blockDim.x + threadIdx.x;
  if (idx >= mtiles * 2048) return;
  int p = idx & 15, lane = (idx >> 4) & 31, kc = (idx >> 9) & 3, mt = idx >> 11;
  int base = (lane & 16) ? 8 : 0;
  int K = kc * 32 + base + (p < 8 ? p : p + 8);
  int row = mt * 16 + (lane & 15);
  dst[((size_t)(mt * kcSlots + kcOff + kc) * 32 + lane) * 16 + p] =
      (_Float16)src[(size_t)row * 128 + K];
}

// ---------------------------------------------------------------------------
// Node projections: A_node = h@We1_a + be1 ; B_node = h@We1_b.  WMMA GEMM.
// 80 blocks x 128 threads (4 waves); each wave owns 4 of the 16 (mat,nt) tiles.
__global__ void k_node_proj(const _Float16* __restrict__ hp,
                            const _Float16* __restrict__ wa,
                            const _Float16* __restrict__ wb,
                            const float* __restrict__ be1,
                            float* __restrict__ An, float* __restrict__ Bn){
  int mt = blockIdx.x;
  int lane = threadIdx.x & 31, wv = threadIdx.x >> 5;
  v8f acc[4] = {};
  #pragma unroll
  for (int kc = 0; kc < 4; ++kc){
    v16h a = *(const v16h*)(hp + ((size_t)(mt * 8 + kc) * 32 + lane) * 16);
    #pragma unroll
    for (int q = 0; q < 4; ++q){
      int pr = wv * 4 + q; int mat = pr >> 3; int nt = pr & 7;
      const _Float16* wp = mat ? wb : wa;
      v16h bf = *(const v16h*)(wp + ((size_t)(kc * 8 + nt) * 32 + lane) * 16);
      acc[q] = WMMA16(a, bf, acc[q]);
    }
  }
  int col0 = lane & 15, half8 = (lane & 16) ? 8 : 0;
  #pragma unroll
  for (int q = 0; q < 4; ++q){
    int pr = wv * 4 + q; int mat = pr >> 3; int nt = pr & 7;
    int c2 = nt * 16 + col0;
    float bias = mat ? 0.f : be1[c2];
    float* dst = mat ? Bn : An;
    const float* ap = (const float*)&acc[q];
    #pragma unroll
    for (int r = 0; r < 8; ++r)
      dst[(size_t)(mt * 16 + r + half8) * HID + c2] = ap[r] + bias;
  }
}

// ---------------------------------------------------------------------------
// Fused edge kernel. Block = (graph bs, 16-row i-tile, j-half of 80). 5 waves,
// one 16-wide j-tile each. Per i-row: build silu(pre1) A-fragments in
// registers, 32 WMMA vs We2 (LDS), fused epilogue: silu+mask, Wx dot (shfl
// butterfly), agg scatter (LDS ds_add_f32), coordinate update via
// x_i*Sum(w) - Sum(x_j*w).  Block partials -> global_atomic_add_f32.
__global__ void __launch_bounds__(EDGE_THREADS)
k_edge(const float* __restrict__ An, const float* __restrict__ Bn,
       const _Float16* __restrict__ we2p, const float* __restrict__ be2,
       const float* __restrict__ wrow,   // We1 rows [2hd..2hd+2]: wd|wb|wt
       const float* __restrict__ Wx, const float* __restrict__ bx,
       const float* __restrict__ t, const float* __restrict__ bond,
       const float* __restrict__ xcur,
       float* __restrict__ agg, float* __restrict__ xout){
  __shared__ __align__(16) float ldsA[16 * 128];
  __shared__ __align__(16) float ldsAgg[16 * 128];
  __shared__ float ldsX[Nn * 3];
  __shared__ __align__(32) _Float16 ldsW2[16384];
  __shared__ float ldsWd[128], ldsWb[128], ldsTw[128], ldsWxv[128], ldsBe2[128];
  __shared__ float ldsXout[48];

  int tid = threadIdx.x;
  int bs = blockIdx.x / 20, rem = blockIdx.x % 20;
  int it = rem >> 1, jh = rem & 1;
  int b = bs >> 2;
  int i0 = it * 16;
  float tv = t[b];

  for (int i = tid; i < 2048; i += EDGE_THREADS)
    ((uint4*)ldsW2)[i] = ((const uint4*)we2p)[i];
  for (int i = tid; i < 2048; i += EDGE_THREADS){
    ldsA[i] = An[(size_t)(bs * Nn + i0) * HID + i];
    ldsAgg[i] = 0.f;
  }
  for (int i = tid; i < Nn * 3; i += EDGE_THREADS)
    ldsX[i] = xcur[(size_t)bs * Nn * 3 + i];
  for (int i = tid; i < 128; i += EDGE_THREADS){
    ldsWd[i] = wrow[i]; ldsWb[i] = wrow[128 + i]; ldsTw[i] = tv * wrow[256 + i];
    ldsWxv[i] = Wx[i];  ldsBe2[i] = be2[i];
  }
  if (tid < 48) ldsXout[tid] = 0.f;
  __syncthreads();

  int wv = tid >> 5, lane = tid & 31;
  int j0 = jh * 80 + wv * 16;
  int jj = lane & 15, half8 = (lane & 16) ? 8 : 0;
  int jrow = j0 + jj;                         // j index of this lane's A rows
  const float* bnp = Bn + (size_t)(bs * Nn + jrow) * HID;
  const float* bondp = bond + (size_t)b * Nn * Nn + jrow;
  float xjx = ldsX[jrow * 3], xjy = ldsX[jrow * 3 + 1], xjz = ldsX[jrow * 3 + 2];
  float bxv = bx[0];
  int kb = (lane & 16) ? 8 : 0;
  float inv = 1.f / (float)(Nn - 1);

  #pragma unroll 1
  for (int ii = 0; ii < 16; ++ii){
    int i = i0 + ii;
    float xix = ldsX[i * 3], xiy = ldsX[i * 3 + 1], xiz = ldsX[i * 3 + 2];
    float dx = xix - xjx, dy = xiy - xjy, dz = xiz - xjz;
    float dist = sqrtf(dx * dx + dy * dy + dz * dz + 1e-12f);
    float bv = bondp[(size_t)i * Nn];
    if (ii + 1 < 16)                 // gfx1250 global_prefetch_b8 for next row
      __builtin_prefetch(&bondp[(size_t)(i + 1) * Nn], 0, 1);

    // Build A-operand fragments of silu(pre1) directly in WMMA layout.
    v16h afr[4];
    #pragma unroll
    for (int kc = 0; kc < 4; ++kc){
      int c0 = kc * 32 + kb;
      Frag16 u;
      #pragma unroll
      for (int p = 0; p < 8; ++p){
        int c = c0 + p;
        float v = ldsA[ii * 128 + c] + bnp[c] + dist * ldsWd[c]
                + bv * ldsWb[c] + ldsTw[c];
        u.h[p] = (_Float16)siluf(v);
        c = c0 + 16 + p;
        v = ldsA[ii * 128 + c] + bnp[c] + dist * ldsWd[c]
          + bv * ldsWb[c] + ldsTw[c];
        u.h[p + 8] = (_Float16)siluf(v);
      }
      afr[kc] = u.v;
    }

    // m2_pre = m1 @ We2 : 8 N-tiles x 4 K-chunks of WMMA.
    v8f acc[8] = {};
    #pragma unroll
    for (int nt = 0; nt < 8; ++nt){
      #pragma unroll
      for (int kc = 0; kc < 4; ++kc){
        v16h bf = *(const v16h*)&ldsW2[((kc * 8 + nt) * 32 + lane) * 16];
        acc[nt] = WMMA16(afr[kc], bf, acc[nt]);
      }
    }

    // Fused epilogue.
    float rowW[8] = {0.f,0.f,0.f,0.f,0.f,0.f,0.f,0.f};
    int col = lane & 15;
    #pragma unroll
    for (int nt = 0; nt < 8; ++nt){
      int c2 = nt * 16 + col;
      float be2v = ldsBe2[c2], wxv = ldsWxv[c2];
      const float* ap = (const float*)&acc[nt];
      float colSum = 0.f;
      #pragma unroll
      for (int r = 0; r < 8; ++r){
        float v = siluf(ap[r] + be2v);
        if (i == j0 + r + half8) v = 0.f;      // edge_mask = 1 - eye
        colSum += v;
        rowW[r] += v * wxv;
      }
      colSum += __shfl_xor(colSum, 16, 32);    // rows 0-7 + rows 8-15
      if (lane < 16) atomicAdd(&ldsAgg[ii * 128 + c2], colSum);
    }
    #pragma unroll
    for (int m = 1; m < 16; m <<= 1)
      #pragma unroll
      for (int r = 0; r < 8; ++r) rowW[r] += __shfl_xor(rowW[r], m, 32);
    float s0 = 0.f, sx = 0.f, sy = 0.f, sz = 0.f;
    #pragma unroll
    for (int r = 0; r < 8; ++r){
      float w = rowW[r] + bxv;
      int j = j0 + r + half8;
      s0 += w;
      sx += ldsX[j * 3] * w; sy += ldsX[j * 3 + 1] * w; sz += ldsX[j * 3 + 2] * w;
    }
    s0 += __shfl_xor(s0, 16, 32); sx += __shfl_xor(sx, 16, 32);
    sy += __shfl_xor(sy, 16, 32); sz += __shfl_xor(sz, 16, 32);
    if (lane == 0){
      atomicAdd(&ldsXout[ii * 3 + 0], xix * s0 - sx);
      atomicAdd(&ldsXout[ii * 3 + 1], xiy * s0 - sy);
      atomicAdd(&ldsXout[ii * 3 + 2], xiz * s0 - sz);
    }
  }
  __syncthreads();
  // Combine j-half partials in global memory (pre-zeroed buffers).
  for (int i = tid; i < 2048; i += EDGE_THREADS)
    atomicAdd(&agg[(size_t)(bs * Nn + i0) * HID + i], ldsAgg[i] * inv);
  if (tid < 48)
    atomicAdd(&xout[(size_t)(bs * Nn + i0) * 3 + tid], ldsXout[tid]);
}

__global__ void k_xupdate(float* __restrict__ xc, const float* __restrict__ xo,
                          const float* __restrict__ mask){
  int idx = blockIdx.x * blockDim.x + threadIdx.x;
  if (idx >= M * 3) return;
  int bs = idx / (Nn * 3), rem = idx % (Nn * 3);
  int b = bs >> 2;
  xc[idx] += xo[idx] * (1.f / (float)(Nn - 1)) * mask[(size_t)b * Nn * 3 + rem];
}

// ---------------------------------------------------------------------------
// Node MLP: h_new = silu([h|agg] @ Wh1 + bh1) @ Wh2 + bh2.  Two WMMA stages,
// intermediate staged + re-packed through LDS.
__global__ void k_node_mlp(const _Float16* __restrict__ hp,
                           const _Float16* __restrict__ wh1p,
                           const _Float16* __restrict__ wh2p,
                           const float* __restrict__ bh1,
                           const float* __restrict__ bh2,
                           float* __restrict__ hout){
  __shared__ __align__(16) _Float16 inter[16 * 128];
  int mt = blockIdx.x;
  int lane = threadIdx.x & 31, wv = threadIdx.x >> 5;
  int col = lane & 15, half8 = (lane & 16) ? 8 : 0;

  v8f acc[2] = {};
  #pragma unroll
  for (int kc = 0; kc < 8; ++kc){
    v16h a = *(const v16h*)(hp + ((size_t)(mt * 8 + kc) * 32 + lane) * 16);
    #pragma unroll
    for (int q = 0; q < 2; ++q){
      int nt = wv * 2 + q;
      v16h bf = *(const v16h*)(wh1p + ((size_t)(kc * 8 + nt) * 32 + lane) * 16);
      acc[q] = WMMA16(a, bf, acc[q]);
    }
  }
  #pragma unroll
  for (int q = 0; q < 2; ++q){
    int nt = wv * 2 + q, c2 = nt * 16 + col;
    const float* ap = (const float*)&acc[q];
    #pragma unroll
    for (int r = 0; r < 8; ++r)
      inter[(r + half8) * 128 + c2] = (_Float16)siluf(ap[r] + bh1[c2]);
  }
  __syncthreads();

  v8f acc2[2] = {};
  #pragma unroll
  for (int kc = 0; kc < 4; ++kc){
    int row = lane & 15;
    int c0 = kc * 32 + ((lane & 16) ? 8 : 0);
    Frag16 u;
    u.q[0] = *(const uint4*)&inter[row * 128 + c0];
    u.q[1] = *(const uint4*)&inter[row * 128 + c0 + 16];
    #pragma unroll
    for (int q = 0; q < 2; ++q){
      int nt = wv * 2 + q;
      v16h bf = *(const v16h*)(wh2p + ((size_t)(kc * 8 + nt) * 32 + lane) * 16);
      acc2[q] = WMMA16(u.v, bf, acc2[q]);
    }
  }
  #pragma unroll
  for (int q = 0; q < 2; ++q){
    int nt = wv * 2 + q, c2 = nt * 16 + col;
    const float* ap = (const float*)&acc2[q];
    #pragma unroll
    for (int r = 0; r < 8; ++r)
      hout[(size_t)(mt * 16 + r + half8) * HID + c2] = ap[r] + bh2[c2];
  }
}

// ---------------------------------------------------------------------------
// pred = (x - x0)*mask, then subtract per-residue mean.
__global__ void k_finalize(const float* __restrict__ xc, const float* __restrict__ x0,
                           const float* __restrict__ mask, const int* __restrict__ apos,
                           float* __restrict__ out){
  __shared__ float sum[NRES * 3];
  __shared__ float cnt[NRES * 3];
  int bs = blockIdx.x, b = bs >> 2, n = threadIdx.x;
  if (n < NRES * 3){ sum[n] = 0.f; cnt[n] = 0.f; }
  __syncthreads();
  int r = apos[b * Nn + n] - 1;
  float p[3];
  #pragma unroll
  for (int d = 0; d < 3; ++d){
    float mk = mask[((size_t)b * Nn + n) * 3 + d];
    p[d] = (xc[((size_t)bs * Nn + n) * 3 + d] - x0[((size_t)bs * Nn + n) * 3 + d]) * mk;
    atomicAdd(&sum[r * 3 + d], p[d]);
    atomicAdd(&cnt[r * 3 + d], mk);
  }
  __syncthreads();
  #pragma unroll
  for (int d = 0; d < 3; ++d){
    float mk = mask[((size_t)b * Nn + n) * 3 + d];
    out[((size_t)bs * Nn + n) * 3 + d] =
        (p[d] - sum[r * 3 + d] / (cnt[r * 3 + d] + 1e-8f)) * mk;
  }
}

// ---------------------------------------------------------------------------
extern "C" void kernel_launch(void* const* d_in, const int* in_sizes, int n_in,
                              void* d_out, int out_size, void* d_ws, size_t ws_size,
                              hipStream_t stream){
  (void)in_sizes; (void)n_in; (void)out_size; (void)ws_size;
  const float* t    = (const float*)d_in[0];
  const float* x    = (const float*)d_in[1];
  const int*   frag = (const int*)d_in[2];
  const int*   atoms= (const int*)d_in[3];
  const int*   apos = (const int*)d_in[4];
  const float* bond = (const float*)d_in[5];
  // d_in[6] edge_mask == 1-eye (self-edge mask hardcoded in k_edge)
  const float* amask= (const float*)d_in[7];
  // params: per-layer dicts flattened alphabetically:
  // We1, We2, Wh1, Wh2, Wx, be1, be2, bh1, bh2, bx
  const float *We1[3], *We2[3], *Wh1[3], *Wh2[3], *Wx[3];
  const float *be1[3], *be2[3], *bh1[3], *bh2[3], *bx[3];
  for (int l = 0; l < 3; ++l){
    int p = 8 + 10 * l;
    We1[l] = (const float*)d_in[p + 0]; We2[l] = (const float*)d_in[p + 1];
    Wh1[l] = (const float*)d_in[p + 2]; Wh2[l] = (const float*)d_in[p + 3];
    Wx [l] = (const float*)d_in[p + 4]; be1[l] = (const float*)d_in[p + 5];
    be2[l] = (const float*)d_in[p + 6]; bh1[l] = (const float*)d_in[p + 7];
    bh2[l] = (const float*)d_in[p + 8]; bx [l] = (const float*)d_in[p + 9];
  }

  char* ws = (char*)d_ws;
  size_t off = 0;
  auto carve = [&](size_t bytes) -> char* {
    char* p = ws + off; off = (off + bytes + 255) & ~(size_t)255; return p;
  };
  float*    h    = (float*)   carve((size_t)M * HID * 4);
  _Float16* hp   = (_Float16*)carve((size_t)MT * 8 * 512 * 2);
  float*    An   = (float*)   carve((size_t)M * HID * 4);
  float*    Bn   = (float*)   carve((size_t)M * HID * 4);
  float*    agg  = (float*)   carve((size_t)M * HID * 4);
  float*    xc   = (float*)   carve((size_t)M * 3 * 4);
  float*    xo   = (float*)   carve((size_t)M * 3 * 4);
  float*    wtmp = (float*)   carve((size_t)256 * 128 * 4);
  _Float16 *wa_p[3], *wb_p[3], *we2_p[3], *wh1_p[3], *wh2_p[3];
  for (int l = 0; l < 3; ++l){
    wa_p [l] = (_Float16*)carve(4 * 8 * 512 * 2);
    wb_p [l] = (_Float16*)carve(4 * 8 * 512 * 2);
    we2_p[l] = (_Float16*)carve(4 * 8 * 512 * 2);
    wh1_p[l] = (_Float16*)carve(8 * 8 * 512 * 2);
    wh2_p[l] = (_Float16*)carve(4 * 8 * 512 * 2);
  }

  k_init <<<(Bb * Nn + 255) / 256, 256, 0, stream>>>(atoms, frag, apos, h);
  k_copyx<<<(M * 3 + 255) / 256, 256, 0, stream>>>(x, xc);
  k_pack_a<<<(MT * 2048 + 255) / 256, 256, 0, stream>>>(h, MT, hp, 8, 0);

  for (int l = 0; l < 3; ++l){
    int hd = l ? 128 : 78;
    // --- weight prep (pad to WMMA K multiples, pack to VGPR layouts) ---
    k_pad2<<<64, 256, 0, stream>>>(We1[l], hd, 0, (const float*)nullptr, 0, 0, wtmp, 128);
    k_pack_b<<<64, 256, 0, stream>>>(wtmp, 4, wa_p[l]);
    k_pad2<<<64, 256, 0, stream>>>(We1[l] + (size_t)hd * 128, hd, 0,
                                   (const float*)nullptr, 0, 0, wtmp, 128);
    k_pack_b<<<64, 256, 0, stream>>>(wtmp, 4, wb_p[l]);
    k_pad2<<<64, 256, 0, stream>>>(We2[l], 128, 0, (const float*)nullptr, 0, 0, wtmp, 128);
    k_pack_b<<<64, 256, 0, stream>>>(wtmp, 4, we2_p[l]);
    k_pad2<<<128, 256, 0, stream>>>(Wh1[l], hd, 0,
                                    Wh1[l] + (size_t)hd * 128, 128, 128, wtmp, 256);
    k_pack_b<<<128, 256, 0, stream>>>(wtmp, 8, wh1_p[l]);
    k_pad2<<<64, 256, 0, stream>>>(Wh2[l], 128, 0, (const float*)nullptr, 0, 0, wtmp, 128);
    k_pack_b<<<64, 256, 0, stream>>>(wtmp, 4, wh2_p[l]);

    // --- layer compute ---
    k_node_proj<<<MT, 128, 0, stream>>>(hp, wa_p[l], wb_p[l], be1[l], An, Bn);
    k_zero<<<(M * HID + 255) / 256, 256, 0, stream>>>(agg, M * HID);
    k_zero<<<(M * 3 + 255) / 256, 256, 0, stream>>>(xo, M * 3);
    k_edge<<<BS * 20, EDGE_THREADS, 0, stream>>>(An, Bn, we2_p[l], be2[l],
                                                 We1[l] + (size_t)2 * hd * 128,
                                                 Wx[l], bx[l], t, bond, xc, agg, xo);
    k_xupdate<<<(M * 3 + 255) / 256, 256, 0, stream>>>(xc, xo, amask);
    k_pack_a<<<(MT * 2048 + 255) / 256, 256, 0, stream>>>(agg, MT, hp, 8, 4);
    k_node_mlp<<<MT, 128, 0, stream>>>(hp, wh1_p[l], wh2_p[l], bh1[l], bh2[l], h);
    k_pack_a<<<(MT * 2048 + 255) / 256, 256, 0, stream>>>(h, MT, hp, 8, 0);
  }

  k_finalize<<<BS, Nn, 0, stream>>>(xc, x, amask, apos, (float*)d_out);
}
